// NodeModel_41283225649617
// MI455X (gfx1250) — matrix operand order repaired and checked by reading, not compiled, not run on hardware
//
#include <hip/hip_runtime.h>
#include <hip/hip_bf16.h>

typedef __attribute__((ext_vector_type(16))) _Float16 v16h;
typedef __attribute__((ext_vector_type(8)))  float    v8f;

union F16x2 { unsigned u; _Float16 h[2]; };
union Frag  { v16h v; unsigned u[8]; };

#define WAVES 4   // 128-thread blocks = 4 wave32

// ---------------------------------------------------------------- zero scratch
__global__ void zero_kernel(float* __restrict__ p, long n) {
    long i = (long)blockIdx.x * blockDim.x + threadIdx.x;
    long stride = (long)gridDim.x * blockDim.x;
    for (; i < n; i += stride) p[i] = 0.0f;
}

// ---------------------------------------------------------------- fused edge MLP + scatter-sum
// per wave: 16 edges. h = relu([x[src],ea] @ w1m + b1m) ; msg = h @ w2m (bias folded into node pass)
// atomicAdd msg into msg_sum[src], atomicAdd 1 into cnt[src]. Scatter addresses are 32-bit element
// offsets (saddr + voffset form); invalid lanes are redirected branch-free to a dump row.
__global__ __launch_bounds__(128) void edge_kernel(
    const float* __restrict__ x, const int* __restrict__ src,
    const float* __restrict__ ea, const float* __restrict__ w1m,
    const float* __restrict__ b1m, const float* __restrict__ w2m,
    float* __restrict__ msg_sum, float* __restrict__ cnt,
    unsigned dumpOff, int E)
{
    __shared__ unsigned Bsw[8192];          // w2m as f16 fragments: [nt][kt][v][lane] -> 32 KB
    __shared__ float    w1s[512];           // w1m (3x128) + b1m (128)
    __shared__ float    stF[WAVES][64];     // per-wave: 16 x (x0,x1,ea,pad)
    __shared__ int      stI[WAVES][16];     // per-wave: src index (-1 = invalid)
    __shared__ _Float16 hA [WAVES][2048];   // per-wave: h tile [16][128] f16 -> 16 KB

    const int tid  = threadIdx.x;
    const int lane = tid & 31;
    const int w    = tid >> 5;
    const int half = lane >> 4;
    const int mN   = lane & 15;

    // one-time per block: stage layer-1 weights + swizzle w2m into B-fragment layout
    for (int i = tid; i < 512; i += blockDim.x)
        w1s[i] = (i < 384) ? w1m[i] : b1m[i - 384];
    for (int i = tid; i < 8192; i += blockDim.x) {
        int l = i & 31, v = (i >> 5) & 7, kt = (i >> 8) & 3, nt = i >> 10;
        int k = kt * 32 + ((l >> 4) << 4) + (v << 1);   // B layout: lanes0-15 K=0..15, lanes16-31 K=16..31
        int n = nt * 16 + (l & 15);
        F16x2 p;
        p.h[0] = (_Float16)w2m[k * 128 + n];
        p.h[1] = (_Float16)w2m[(k + 1) * 128 + n];
        Bsw[i] = p.u;
    }
    __syncthreads();

    const long gridWaves = (long)gridDim.x * WAVES;
    const long gw        = (long)blockIdx.x * WAVES + w;
    const int  numTiles  = (E + 15) >> 4;
    const int  iters     = (int)((numTiles + gridWaves - 1) / gridWaves);

    for (int it = 0; it < iters; ++it) {
        long tile   = gw + (long)it * gridWaves;
        bool active = tile < numTiles;                 // wave-uniform
        long base   = tile << 4;

        if (active && lane < 16) {                     // stage 16 edges
            long e = base + lane;
            int s = -1; float x0 = 0.f, x1 = 0.f, ev = 0.f;
            if (e < (long)E) {
                s  = src[e];
                x0 = x[2 * s]; x1 = x[2 * s + 1]; ev = ea[e];
                atomicAdd(&cnt[s], 1.0f);
            }
            stF[w][lane * 4 + 0] = x0;
            stF[w][lane * 4 + 1] = x1;
            stF[w][lane * 4 + 2] = ev;
            stI[w][lane] = s;
        }
        __syncthreads();

        if (active) {                                  // layer 1: K=3 VALU -> f16 tile
            #pragma unroll
            for (int t = 0; t < 4; ++t) {
                int j = lane + (t << 5);
                float W0 = w1s[j], W1 = w1s[128 + j], W2 = w1s[256 + j], B = w1s[384 + j];
                #pragma unroll
                for (int m = 0; m < 16; ++m) {
                    float h = fmaf(stF[w][m * 4], W0,
                              fmaf(stF[w][m * 4 + 1], W1,
                              fmaf(stF[w][m * 4 + 2], W2, B)));
                    hA[w][m * 128 + j] = (_Float16)(h > 0.f ? h : 0.f);
                }
            }
        }
        __syncthreads();

        if (active) {                                  // layer 2: WMMA f16 -> f32, then atomic scatter
            // 32-bit scatter offsets (branch-free dump-row redirect for invalid rows)
            unsigned off[8];
            #pragma unroll
            for (int v = 0; v < 8; ++v) {
                int mr = v + (half << 3);
                int s  = stI[w][mr];
                unsigned o = (s >= 0) ? (unsigned)s * 128u : dumpOff;
                off[v] = o + (unsigned)mN;
            }

            const unsigned* hAu = (const unsigned*)&hA[w][0];
            Frag a[4];
            #pragma unroll
            for (int kt = 0; kt < 4; ++kt)
                #pragma unroll
                for (int v = 0; v < 8; ++v) {          // A layout: lanes0-15 K=0..15, lanes16-31 K=16..31
                    int k = kt * 32 + (half << 4) + (v << 1);
                    a[kt].u[v] = hAu[mN * 64 + (k >> 1)];
                }

            #pragma unroll
            for (int nt = 0; nt < 8; ++nt) {
                v8f acc = {};
                #pragma unroll
                for (int kt = 0; kt < 4; ++kt) {
                    Frag b;
                    const unsigned* bp = &Bsw[((nt * 4 + kt) << 8) + lane];
                    #pragma unroll
                    for (int v = 0; v < 8; ++v) b.u[v] = bp[v << 5];
                    acc = __builtin_amdgcn_wmma_f32_16x16x32_f16(
                        false, a[kt].v, false, b.v, (short)0, acc, false, false);
                }
                #pragma unroll
                for (int v = 0; v < 8; ++v)            // saddr + voffset, imm = nt*64B
                    atomicAdd(&msg_sum[off[v] + (unsigned)(nt * 16)], acc[v]);
            }
        }
        __syncthreads();
    }
}

// ---------------------------------------------------------------- node MLP
// agg = msg_sum/cnt + b2m (0 if cnt==0); h2 = relu([x,agg] @ w1n + b1n) via WMMA (agg part)
// with x/bias folded into C; out = h2 @ w2n + b2n by VALU.
__global__ __launch_bounds__(128) void node_kernel(
    const float* __restrict__ x, const float* __restrict__ w1n,
    const float* __restrict__ b1n, const float* __restrict__ w2n,
    const float* __restrict__ b2n, const float* __restrict__ b2m,
    const float* __restrict__ msg_sum, const float* __restrict__ cnt,
    float* __restrict__ out, int N)
{
    __shared__ unsigned Bsw[8192];          // w1n rows 2..129 as B fragments -> 32 KB
    __shared__ float    wss[770];           // [0:128) w1n row0 | [128:256) row1 | [256:384) b1n
                                            // [384:512) b2m | [512:768) w2n | [768:770) b2n
    __shared__ float    stF[WAVES][64];     // 16 x (x0,x1,inv,biasEn)
    __shared__ int      stI[WAVES][16];
    __shared__ _Float16 hA [WAVES][2048];   // agg tile f16; reused for h2 f16

    const int tid  = threadIdx.x;
    const int lane = tid & 31;
    const int w    = tid >> 5;
    const int half = lane >> 4;
    const int mN   = lane & 15;

    for (int i = tid; i < 770; i += blockDim.x) {
        float v;
        if      (i < 256) v = w1n[i];           // rows 0,1 of w1n
        else if (i < 384) v = b1n[i - 256];
        else if (i < 512) v = b2m[i - 384];
        else if (i < 768) v = w2n[i - 512];
        else              v = b2n[i - 768];
        wss[i] = v;
    }
    for (int i = tid; i < 8192; i += blockDim.x) {
        int l = i & 31, v = (i >> 5) & 7, kt = (i >> 8) & 3, nt = i >> 10;
        int k = kt * 32 + ((l >> 4) << 4) + (v << 1);
        int n = nt * 16 + (l & 15);
        F16x2 p;
        p.h[0] = (_Float16)w1n[(2 + k) * 128 + n];
        p.h[1] = (_Float16)w1n[(3 + k) * 128 + n];
        Bsw[i] = p.u;
    }
    __syncthreads();

    const long gridWaves = (long)gridDim.x * WAVES;
    const long gw        = (long)blockIdx.x * WAVES + w;
    const int  numTiles  = (N + 15) >> 4;
    const int  iters     = (int)((numTiles + gridWaves - 1) / gridWaves);

    for (int it = 0; it < iters; ++it) {
        long tile   = gw + (long)it * gridWaves;
        bool active = tile < numTiles;                 // wave-uniform
        long base   = tile << 4;

        if (active && lane < 16) {
            long i = base + lane;
            int id = -1; float x0 = 0.f, x1 = 0.f, inv = 0.f, ben = 0.f;
            if (i < (long)N) {
                id = (int)i;
                x0 = x[2 * i]; x1 = x[2 * i + 1];
                float c = cnt[i];
                if (c > 0.f) { inv = 1.0f / c; ben = 1.0f; }
            }
            stF[w][lane * 4 + 0] = x0;
            stF[w][lane * 4 + 1] = x1;
            stF[w][lane * 4 + 2] = inv;
            stF[w][lane * 4 + 3] = ben;
            stI[w][lane] = id;
        }
        __syncthreads();

        if (active) {                                  // build agg tile (f16), 32-bit gather offsets
            #pragma unroll
            for (int m = 0; m < 16; ++m) {
                int   im  = stI[w][m];
                float inv = stF[w][m * 4 + 2];
                float ben = stF[w][m * 4 + 3];
                unsigned mo = (unsigned)(im < 0 ? 0 : im) * 128u;
                #pragma unroll
                for (int t = 0; t < 4; ++t) {
                    int k = lane + (t << 5);
                    float val = 0.f;
                    if (im >= 0)
                        val = fmaf(msg_sum[mo + (unsigned)k], inv, ben * wss[384 + k]);
                    hA[w][m * 128 + k] = (_Float16)val;
                }
            }
        }
        __syncthreads();

        if (active) {                                  // WMMA layer 1 (agg part), x/bias in C
            const unsigned* hAu = (const unsigned*)&hA[w][0];
            Frag a[4];
            #pragma unroll
            for (int kt = 0; kt < 4; ++kt)
                #pragma unroll
                for (int v = 0; v < 8; ++v) {
                    int k = kt * 32 + (half << 4) + (v << 1);
                    a[kt].u[v] = hAu[mN * 64 + (k >> 1)];
                }
            __builtin_amdgcn_wave_barrier();           // A reads before hA reuse below

            #pragma unroll
            for (int nt = 0; nt < 8; ++nt) {
                int n = nt * 16 + mN;
                v8f acc = {};
                #pragma unroll
                for (int v = 0; v < 8; ++v) {          // C init: b1n + x0*w1n[0] + x1*w1n[1]
                    int mr = v + (half << 3);
                    acc[v] = fmaf(stF[w][mr * 4], wss[n],
                             fmaf(stF[w][mr * 4 + 1], wss[128 + n], wss[256 + n]));
                }
                #pragma unroll
                for (int kt = 0; kt < 4; ++kt) {
                    Frag b;
                    const unsigned* bp = &Bsw[((nt * 4 + kt) << 8) + lane];
                    #pragma unroll
                    for (int v = 0; v < 8; ++v) b.u[v] = bp[v << 5];
                    acc = __builtin_amdgcn_wmma_f32_16x16x32_f16(
                        false, a[kt].v, false, b.v, (short)0, acc, false, false);
                }
                #pragma unroll
                for (int v = 0; v < 8; ++v) {          // relu -> h2 tile (reuse hA)
                    int mr = v + (half << 3);
                    float h = acc[v];
                    hA[w][mr * 128 + n] = (_Float16)(h > 0.f ? h : 0.f);
                }
            }
        }
        __syncthreads();

        if (active) {                                  // layer 2: 128 -> 2, lane = (node, col)
            int mr = lane & 15;
            int c  = lane >> 4;
            int im = stI[w][mr];
            if (im >= 0) {
                float o = wss[768 + c];
                #pragma unroll 8
                for (int jj = 0; jj < 128; ++jj) {
                    int j = (jj + (mr << 3)) & 127;    // rotate to avoid LDS bank conflicts
                    o = fmaf((float)hA[w][mr * 128 + j], wss[512 + j * 2 + c], o);
                }
                out[(long)im * 2 + c] = o;
            }
        }
        __syncthreads();
    }
}

// ---------------------------------------------------------------- launch
extern "C" void kernel_launch(void* const* d_in, const int* in_sizes, int n_in,
                              void* d_out, int out_size, void* d_ws, size_t ws_size,
                              hipStream_t stream) {
    const float* x          = (const float*)d_in[0];
    const int*   edge_index = (const int*)  d_in[1];
    const float* edge_attr  = (const float*)d_in[2];
    const float* w1m        = (const float*)d_in[5];
    const float* b1m        = (const float*)d_in[6];
    const float* w2m        = (const float*)d_in[7];
    const float* b2m        = (const float*)d_in[8];
    const float* w1n        = (const float*)d_in[9];
    const float* b1n        = (const float*)d_in[10];
    const float* w2n        = (const float*)d_in[11];
    const float* b2n        = (const float*)d_in[12];

    const int N = in_sizes[0] / 2;      // 100000
    const int E = in_sizes[2];          // 3200000
    const int* src = edge_index + E;    // edge_index[1] row

    float* msg_sum   = (float*)d_ws;                     // [N,128]
    float* cnt       = msg_sum + (size_t)N * 128;        // [N]
    unsigned dumpOff = (unsigned)N * 128u + (unsigned)N; // dump row right after cnt

    long zn = (long)N * 128 + N + 128;
    zero_kernel<<<1024, 256, 0, stream>>>(msg_sum, zn);
    edge_kernel<<<1024, 128, 0, stream>>>(x, src, edge_attr, w1m, b1m, w2m,
                                          msg_sum, cnt, dumpOff, E);
    node_kernel<<<512, 128, 0, stream>>>(x, w1n, b1n, w2n, b2n, b2m,
                                         msg_sum, cnt, (float*)d_out, N);
}
